// DynamicWeightedAverage_73358041416238
// MI455X (gfx1250) — matrix operand order repaired and checked by reading, not compiled, not run on hardware
//
#include <hip/hip_runtime.h>
#include <hip/hip_bf16.h>
#include <math.h>

typedef __attribute__((ext_vector_type(16))) _Float16 v16h;
typedef __attribute__((ext_vector_type(2)))  _Float16 h2;
typedef __attribute__((ext_vector_type(8)))  float    v8f;

#define DIMN    128
#define W1PITCH 130      // 128 + 2 halfs pad -> 260B row stride, conflict-free
#define NBLK    640      // persistent blocks for score / sumexp kernels
#define TPB     256      // 8 waves (wave32)

// -------------------------------------------------------------------------
// Kernel 1: exclusive scan of lengths -> offsets[B+1]
// -------------------------------------------------------------------------
__global__ __launch_bounds__(1024) void dwa_scan_kernel(
    const int* __restrict__ lengths, int* __restrict__ offsets, int B)
{
  __shared__ int tmp[1024];
  __shared__ int carry;
  if (threadIdx.x == 0) { carry = 0; offsets[0] = 0; }
  __syncthreads();
  for (int base = 0; base < B; base += 1024) {
    int i = base + (int)threadIdx.x;
    int v = (i < B) ? lengths[i] : 0;
    tmp[threadIdx.x] = v;
    __syncthreads();
    for (int off = 1; off < 1024; off <<= 1) {
      int t = (threadIdx.x >= (unsigned)off) ? tmp[threadIdx.x - off] : 0;
      __syncthreads();
      tmp[threadIdx.x] += t;
      __syncthreads();
    }
    if (i < B) offsets[i + 1] = carry + tmp[threadIdx.x];
    __syncthreads();
    if (threadIdx.x == 0) carry += tmp[1023];
    __syncthreads();
  }
}

// -------------------------------------------------------------------------
// Kernel 2: scores[t] = relu(emb[t] @ W1 + b1) @ W2 + b2  (f16 WMMA, f32 acc)
// One wave per 16-row tile; h tile (16x128) lives entirely in accumulators.
// Also produces per-block max of the scores it generated.
// -------------------------------------------------------------------------
__global__ __launch_bounds__(TPB) void dwa_score_kernel(
    const float* __restrict__ emb, const float* __restrict__ W1,
    const float* __restrict__ b1,  const float* __restrict__ W2,
    const float* __restrict__ b2,  float* __restrict__ scores,
    float* __restrict__ blockmax,  int T)
{
  __shared__ _Float16 w1s[DIMN * W1PITCH];
  __shared__ float redmax[TPB / 32];

  // Convert W1 (f32, row-major [k][col]) -> LDS f16 [col][k] padded
  for (int e = threadIdx.x; e < DIMN * DIMN; e += blockDim.x) {
    int k = e >> 7, col = e & (DIMN - 1);
    w1s[col * W1PITCH + k] = (_Float16)W1[e];
  }
  __syncthreads();

  const int lane = threadIdx.x & 31;
  const int wave = threadIdx.x >> 5;
  const int g    = lane >> 4;     // lane half (0/1)
  const int r    = lane & 15;

  // Per-lane column constants (column N = 16*n + r for both halves)
  float b1c[8], w2c[8];
#pragma unroll
  for (int n = 0; n < 8; ++n) { b1c[n] = b1[n * 16 + r]; w2c[n] = W2[n * 16 + r]; }
  const float bias2 = b2[0];

  const int totalWaves = (int)(gridDim.x * (blockDim.x >> 5));
  const int waveId     = (int)blockIdx.x * (int)(blockDim.x >> 5) + wave;
  const int nTiles     = T >> 4;
  float localmax = -INFINITY;

  for (int tile = waveId; tile < nTiles; tile += totalWaves) {
    const int row0 = tile << 4;
    v8f acc[8];
#pragma unroll
    for (int n = 0; n < 8; ++n) acc[n] = (v8f){0.f,0.f,0.f,0.f,0.f,0.f,0.f,0.f};

#pragma unroll
    for (int ks = 0; ks < 4; ++ks) {
      // ---- A fragment: 16x32 f16. Lane(r,g) holds row r, K in
      //      {ks*32 + g*8 + [0..7]} U {ks*32 + g*8 + [16..23]}
      const float* src = emb + (size_t)(row0 + r) * DIMN + ks * 32 + g * 8;
      v16h afrag;
#pragma unroll
      for (int j = 0; j < 8; ++j) {
        int kloc = ((j >> 2) << 4) + ((j & 3) << 1);
        float2 f = *(const float2*)(src + kloc);
        afrag[2 * j]     = (_Float16)f.x;
        afrag[2 * j + 1] = (_Float16)f.y;
      }
#pragma unroll
      for (int n = 0; n < 8; ++n) {
        // ---- B fragment: 32x16 f16, column N = 16n + r,
        //      K in {ks*32 + g*8 + [0..7]} U {ks*32 + g*8 + [16..23]}.
        const _Float16* bs = &w1s[(n * 16 + r) * W1PITCH + ks * 32 + g * 8];
        v16h bfrag;
#pragma unroll
        for (int j = 0; j < 8; ++j) {
          h2 hv = *(const h2*)(bs + ((j >> 2) << 4) + ((j & 3) << 1));       // 32-bit aligned ds load
          bfrag[2 * j]     = hv.x;
          bfrag[2 * j + 1] = hv.y;
        }
        acc[n] = __builtin_amdgcn_wmma_f32_16x16x32_f16(
            false, afrag, false, bfrag, (short)0, acc[n], false, false);
      }
    }

    // ---- Epilogue: bias + ReLU + dot(W2); acc[n][j] = h[row0+j+8g][16n+r]
    float part[8];
#pragma unroll
    for (int j = 0; j < 8; ++j) {
      float s = 0.f;
#pragma unroll
      for (int n = 0; n < 8; ++n) {
        float hv = acc[n][j] + b1c[n];
        hv = hv > 0.f ? hv : 0.f;
        s += hv * w2c[n];
      }
      part[j] = s;
    }
    // Reduce across the 16 lanes of each half (columns 0..127)
#pragma unroll
    for (int j = 0; j < 8; ++j) {
      float s = part[j];
      for (int off = 8; off >= 1; off >>= 1) s += __shfl_xor(s, off, 16);
      part[j] = s;
    }
    if (r == 0) {
#pragma unroll
      for (int j = 0; j < 8; ++j) {
        float sc = part[j] + bias2;
        scores[row0 + g * 8 + j] = sc;
        localmax = fmaxf(localmax, sc);
      }
    }
  }

  // Block max reduction (deterministic)
  for (int off = 16; off >= 1; off >>= 1)
    localmax = fmaxf(localmax, __shfl_xor(localmax, off, 32));
  if (lane == 0) redmax[wave] = localmax;
  __syncthreads();
  if (threadIdx.x == 0) {
    float m = redmax[0];
    for (int w = 1; w < (int)(blockDim.x >> 5); ++w) m = fmaxf(m, redmax[w]);
    blockmax[blockIdx.x] = m;
  }
}

// -------------------------------------------------------------------------
// Kernel 3: reduce per-block maxima -> stats[0]
// -------------------------------------------------------------------------
__global__ __launch_bounds__(1024) void dwa_maxred_kernel(
    const float* __restrict__ blockmax, float* __restrict__ stats, int n)
{
  __shared__ float red[1024];
  float m = -INFINITY;
  for (int i = threadIdx.x; i < n; i += blockDim.x) m = fmaxf(m, blockmax[i]);
  red[threadIdx.x] = m;
  __syncthreads();
  for (int off = 512; off >= 1; off >>= 1) {
    if (threadIdx.x < (unsigned)off)
      red[threadIdx.x] = fmaxf(red[threadIdx.x], red[threadIdx.x + off]);
    __syncthreads();
  }
  if (threadIdx.x == 0) stats[0] = red[0];
}

// -------------------------------------------------------------------------
// Kernel 4: per-block partial sums of exp(score - gmax)
// -------------------------------------------------------------------------
__global__ __launch_bounds__(TPB) void dwa_sumexp_kernel(
    const float* __restrict__ scores, const float* __restrict__ stats,
    float* __restrict__ blocksum, int T)
{
  __shared__ float red[TPB / 32];
  const float gmax = stats[0];
  float s = 0.f;
  for (int i = (int)(blockIdx.x * blockDim.x + threadIdx.x); i < T;
       i += (int)(gridDim.x * blockDim.x))
    s += expf(scores[i] - gmax);
  for (int off = 16; off >= 1; off >>= 1) s += __shfl_xor(s, off, 32);
  int lane = threadIdx.x & 31, wave = threadIdx.x >> 5;
  if (lane == 0) red[wave] = s;
  __syncthreads();
  if (threadIdx.x == 0) {
    float t = 0.f;
    for (int w = 0; w < (int)(blockDim.x >> 5); ++w) t += red[w];
    blocksum[blockIdx.x] = t;
  }
}

// -------------------------------------------------------------------------
// Kernel 5: reduce per-block sums -> stats[1]
// -------------------------------------------------------------------------
__global__ __launch_bounds__(1024) void dwa_sumred_kernel(
    const float* __restrict__ blocksum, float* __restrict__ stats, int n)
{
  __shared__ float red[1024];
  float s = 0.f;
  for (int i = threadIdx.x; i < n; i += blockDim.x) s += blocksum[i];
  red[threadIdx.x] = s;
  __syncthreads();
  for (int off = 512; off >= 1; off >>= 1) {
    if (threadIdx.x < (unsigned)off) red[threadIdx.x] += red[threadIdx.x + off];
    __syncthreads();
  }
  if (threadIdx.x == 0) stats[1] = red[0];
}

// -------------------------------------------------------------------------
// Kernel 6: pooled[b,d] = sum_{t in segment b} emb[t,d] * softmax_weight[t]
// One block per segment, 128 threads (one per dim). Weights via LDS.
// -------------------------------------------------------------------------
__global__ __launch_bounds__(DIMN) void dwa_pool_kernel(
    const float* __restrict__ emb, const float* __restrict__ scores,
    const int* __restrict__ offsets, const float* __restrict__ stats,
    float* __restrict__ out)
{
  __shared__ float wsm[DIMN];
  const int b = blockIdx.x;
  const int d = threadIdx.x;
  const int start = offsets[b];
  const int end   = offsets[b + 1];
  const float gmax = stats[0];
  const float inv  = 1.0f / stats[1];

  float acc = 0.f;
  for (int base = start; base < end; base += DIMN) {
    __syncthreads();
    int t = base + d;
    if (t < end) wsm[d] = expf(scores[t] - gmax) * inv;
    __syncthreads();
    int cnt = min(DIMN, end - base);
    for (int i = 0; i < cnt; ++i)
      acc += emb[(size_t)(base + i) * DIMN + d] * wsm[i];
  }
  out[(size_t)b * DIMN + d] = acc;
}

// -------------------------------------------------------------------------
extern "C" void kernel_launch(void* const* d_in, const int* in_sizes, int n_in,
                              void* d_out, int out_size, void* d_ws, size_t ws_size,
                              hipStream_t stream)
{
  const float* emb     = (const float*)d_in[0];
  const float* W1      = (const float*)d_in[1];
  const float* b1      = (const float*)d_in[2];
  const float* W2      = (const float*)d_in[3];
  const float* b2      = (const float*)d_in[4];
  const int*   lengths = (const int*)d_in[5];

  const int T = in_sizes[0] / DIMN;   // 819200
  const int B = in_sizes[5];          // 16384
  float* out = (float*)d_out;

  // Workspace layout (~3.4 MB)
  float* scores   = (float*)d_ws;           // T
  float* stats    = scores + T;             // 2: [gmax, sumexp]
  float* blockmax = stats + 2;              // NBLK
  float* blocksum = blockmax + NBLK;        // NBLK
  int*   offsets  = (int*)(blocksum + NBLK); // B+1

  dwa_scan_kernel  <<<1,    1024, 0, stream>>>(lengths, offsets, B);
  dwa_score_kernel <<<NBLK, TPB,  0, stream>>>(emb, W1, b1, W2, b2,
                                               scores, blockmax, T);
  dwa_maxred_kernel<<<1,    1024, 0, stream>>>(blockmax, stats, NBLK);
  dwa_sumexp_kernel<<<NBLK, TPB,  0, stream>>>(scores, stats, blocksum, T);
  dwa_sumred_kernel<<<1,    1024, 0, stream>>>(blocksum, stats, NBLK);
  dwa_pool_kernel  <<<B,    DIMN, 0, stream>>>(emb, scores, offsets, stats, out);
}
